// Rwkv7TimeMixing_43963285242490
// MI455X (gfx1250) — compile-verified, matrix operand order checked by reading, not run on hardware
//
#include <hip/hip_runtime.h>
#include <hip/hip_bf16.h>

// ---------------------------------------------------------------------------
// RWKV7 time-mixing for MI455X (gfx1250, wave32, WMMA).
// Pipeline: weight-cvt -> mix -> bf16 WMMA GEMMs -> act -> WMMA LoRA-up ->
//           prep (in-place) -> WKV7 scan (4 lanes/state-row) -> gn/gate ->
//           Wo WMMA GEMM.
// Big GEMMs (N%128==0): 64x64 per wave (16 WMMA / 16 loads per K-step).
// ---------------------------------------------------------------------------

typedef __attribute__((ext_vector_type(16))) __bf16 bf16x16;
typedef __attribute__((ext_vector_type(8)))  __bf16 bf16x8;
typedef __attribute__((ext_vector_type(8)))  float  f32x8;

#define DEV __device__ __forceinline__

DEV __bf16 f2bf(float f) {
  union { float f; unsigned u; } v; v.f = f;
  unsigned r = v.u + 0x7FFFu + ((v.u >> 16) & 1u);   // round-to-nearest-even
  unsigned short h = (unsigned short)(r >> 16);
  __bf16 b; __builtin_memcpy(&b, &h, sizeof(b));
  return b;
}

DEV float sigmoidf_(float x) { return 1.0f / (1.0f + __expf(-x)); }

DEV float wave_sum(float x) {
#pragma unroll
  for (int off = 16; off > 0; off >>= 1) x += __shfl_xor(x, off, 32);
  return x;
}

#define BF16CAT(lo, hi) \
  __builtin_shufflevector(lo, hi, 0,1,2,3,4,5,6,7,8,9,10,11,12,13,14,15)

// ---------------------------------------------------------------------------
// f32 -> bf16 copy (weight conversion)
// ---------------------------------------------------------------------------
__global__ __launch_bounds__(256) void cvt_bf16_kernel(
    const float* __restrict__ in, __bf16* __restrict__ out, int n) {
  int i = blockIdx.x * 256 + threadIdx.x;
  if (i < n) out[i] = f2bf(in[i]);
}

// ---------------------------------------------------------------------------
// activation + f32 -> bf16  (mode: 0=copy, 1=tanh, 2=sigmoid)
// ---------------------------------------------------------------------------
__global__ __launch_bounds__(256) void act_cvt_kernel(
    const float* __restrict__ in, __bf16* __restrict__ out, int n, int mode) {
  int i = blockIdx.x * 256 + threadIdx.x;
  if (i >= n) return;
  float v = in[i];
  if (mode == 1) v = tanhf(v);
  else if (mode == 2) v = sigmoidf_(v);
  out[i] = f2bf(v);
}

// ---------------------------------------------------------------------------
// token shift + six mixes, written as bf16 GEMM operands
// ---------------------------------------------------------------------------
__global__ __launch_bounds__(256) void mix_kernel(
    const float* __restrict__ x,
    const float* __restrict__ mr, const float* __restrict__ mw,
    const float* __restrict__ mk, const float* __restrict__ mv,
    const float* __restrict__ ma, const float* __restrict__ mg,
    __bf16* __restrict__ xr, __bf16* __restrict__ xw,
    __bf16* __restrict__ xk, __bf16* __restrict__ xv,
    __bf16* __restrict__ xa, __bf16* __restrict__ xg,
    int total, int D, int L) {
  int idx = blockIdx.x * 256 + threadIdx.x;
  if (idx >= total) return;
  int d = idx & (D - 1);
  int token = idx / D;
  float xc = x[idx];
  float xp = ((token % L) == 0) ? 0.0f : x[idx - D];
  float xx = xp - xc;
  xr[idx] = f2bf(xc + xx * mr[d]);
  xw[idx] = f2bf(xc + xx * mw[d]);
  xk[idx] = f2bf(xc + xx * mk[d]);
  xv[idx] = f2bf(xc + xx * mv[d]);
  xa[idx] = f2bf(xc + xx * ma[d]);
  xg[idx] = f2bf(xc + xx * mg[d]);
}

// ---------------------------------------------------------------------------
// WMMA fragment loader: both operands are row-major K-contiguous (x @ W.T),
// so A and B fragments use the identical per-lane pattern.
// ---------------------------------------------------------------------------
DEV bf16x16 load_frag(const __bf16* p) {
  bf16x8 lo = *(const bf16x8*)(p);
  bf16x8 hi = *(const bf16x8*)(p + 16);
  return BF16CAT(lo, hi);
}

// ---------------------------------------------------------------------------
// Small WMMA GEMM: wave tile 32x32 (2x2), block 128x64. For LoRA-down
// (N = 32..128). Requires M%128==0, N%32==0, K%32==0.
// ---------------------------------------------------------------------------
__global__ __launch_bounds__(256) void gemm_bf16_kernel(
    const __bf16* __restrict__ A, const __bf16* __restrict__ Bw,
    float* __restrict__ C, int M, int N, int K) {
  const int lane  = threadIdx.x & 31;
  const int wid   = threadIdx.x >> 5;
  const int m0 = blockIdx.x * 128 + (wid >> 1) * 32;
  const int n0 = blockIdx.y * 64  + (wid & 1) * 32;
  if (n0 >= N) return;  // wave-uniform: EXEC stays all-ones around WMMA

  const int lrow  = lane & 15;
  const int khalf = (lane >> 4) << 3;  // 0 or 8

  const __bf16* aptr0 = A  + (size_t)(m0 + lrow) * K + khalf;
  const __bf16* aptr1 = A  + (size_t)(m0 + 16 + lrow) * K + khalf;
  const __bf16* bptr0 = Bw + (size_t)(n0 + lrow) * K + khalf;
  const __bf16* bptr1 = Bw + (size_t)(n0 + 16 + lrow) * K + khalf;

  f32x8 c00 = {}, c01 = {}, c10 = {}, c11 = {};

  for (int k = 0; k < K; k += 32) {
    bf16x16 a0 = load_frag(aptr0 + k);
    bf16x16 a1 = load_frag(aptr1 + k);
    bf16x16 b0 = load_frag(bptr0 + k);
    bf16x16 b1 = load_frag(bptr1 + k);
    c00 = __builtin_amdgcn_wmma_f32_16x16x32_bf16(false, a0, false, b0,
                                                  (short)0, c00, false, false);
    c01 = __builtin_amdgcn_wmma_f32_16x16x32_bf16(false, a0, false, b1,
                                                  (short)0, c01, false, false);
    c10 = __builtin_amdgcn_wmma_f32_16x16x32_bf16(false, a1, false, b0,
                                                  (short)0, c10, false, false);
    c11 = __builtin_amdgcn_wmma_f32_16x16x32_bf16(false, a1, false, b1,
                                                  (short)0, c11, false, false);
  }

  const int crow = (lane >> 4) << 3;  // 0 or 8
  const int ccol = lane & 15;
#pragma unroll
  for (int e = 0; e < 8; ++e) {
    size_t r0 = (size_t)(m0 + crow + e) * N;
    size_t r1 = (size_t)(m0 + 16 + crow + e) * N;
    C[r0 + n0 + ccol]      = c00[e];
    C[r0 + n0 + 16 + ccol] = c01[e];
    C[r1 + n0 + ccol]      = c10[e];
    C[r1 + n0 + 16 + ccol] = c11[e];
  }
}

// ---------------------------------------------------------------------------
// Big WMMA GEMM: wave tile 64x64 (4x4 tiles -> 16 WMMA per K-step against
// 16 b128 loads: 32 FLOP/byte of operand traffic). Block: 8 waves arranged
// 4(M) x 2(N) -> 256x128. Requires M%256==0, N%128==0, K%32==0.
// ---------------------------------------------------------------------------
__global__ __launch_bounds__(256) void gemm_bf16_big_kernel(
    const __bf16* __restrict__ A, const __bf16* __restrict__ Bw,
    float* __restrict__ C, int M, int N, int K) {
  const int lane = threadIdx.x & 31;
  const int wid  = threadIdx.x >> 5;
  const int m0 = blockIdx.x * 256 + (wid >> 1) * 64;
  const int n0 = blockIdx.y * 128 + (wid & 1) * 64;

  const int lrow  = lane & 15;
  const int khalf = (lane >> 4) << 3;  // 0 or 8

  const __bf16* ap[4];
  const __bf16* bp[4];
#pragma unroll
  for (int i = 0; i < 4; ++i) {
    ap[i] = A  + (size_t)(m0 + 16 * i + lrow) * K + khalf;
    bp[i] = Bw + (size_t)(n0 + 16 * i + lrow) * K + khalf;
  }

  f32x8 c[4][4];
#pragma unroll
  for (int i = 0; i < 4; ++i)
#pragma unroll
    for (int j = 0; j < 4; ++j) c[i][j] = (f32x8){};

  for (int k = 0; k < K; k += 32) {
    __builtin_prefetch(ap[0] + k + 64, 0, 1);
    __builtin_prefetch(bp[0] + k + 64, 0, 1);
    bf16x16 af[4], bf[4];
#pragma unroll
    for (int i = 0; i < 4; ++i) af[i] = load_frag(ap[i] + k);
#pragma unroll
    for (int j = 0; j < 4; ++j) bf[j] = load_frag(bp[j] + k);
#pragma unroll
    for (int i = 0; i < 4; ++i)
#pragma unroll
      for (int j = 0; j < 4; ++j)
        c[i][j] = __builtin_amdgcn_wmma_f32_16x16x32_bf16(
            false, af[i], false, bf[j], (short)0, c[i][j], false, false);
  }

  const int crow = (lane >> 4) << 3;  // 0 or 8
  const int ccol = lane & 15;
#pragma unroll
  for (int i = 0; i < 4; ++i) {
#pragma unroll
    for (int e = 0; e < 8; ++e) {
      size_t rr = (size_t)(m0 + 16 * i + crow + e) * N;
#pragma unroll
      for (int j = 0; j < 4; ++j)
        C[rr + n0 + 16 * j + ccol] = c[i][j][e];
    }
  }
}

// ---------------------------------------------------------------------------
// prep: one wave per (b,l,h) vector of 64. Rewrites scan operands IN PLACE:
//   kf := k*(1+(iclr-1)*k_a), vf := v + (v_first - v)*vv,
//   wf := exp(-0.606531*sigmoid(w_pre+w_b)), af := -kk, bf := kk*iclr
// ---------------------------------------------------------------------------
__global__ __launch_bounds__(256) void prep_kernel(
    float* __restrict__ kf, float* __restrict__ vf, float* __restrict__ wf,
    float* __restrict__ af, float* __restrict__ bf,
    const float* __restrict__ v_first,
    const float* __restrict__ k_k, const float* __restrict__ k_a,
    const float* __restrict__ w_bias, const float* __restrict__ v_bias,
    const float* __restrict__ a_bias, int nvec, int H) {
  int lane = threadIdx.x & 31;
  int vec  = blockIdx.x * 8 + (threadIdx.x >> 5);
  if (vec >= nvec) return;
  int h = vec & (H - 1);
  size_t base = (size_t)vec * 64;
  int e0 = lane, e1 = lane + 32;
  int d0 = h * 64 + e0, d1 = h * 64 + e1;

  float kr0 = kf[base + e0], kr1 = kf[base + e1];
  float kkn0 = kr0 * k_k[d0], kkn1 = kr1 * k_k[d1];
  float ss  = wave_sum(kkn0 * kkn0 + kkn1 * kkn1);
  float inv = 1.0f / fmaxf(sqrtf(ss), 1e-7f);
  float kk0 = kkn0 * inv, kk1 = kkn1 * inv;

  float iclr0 = sigmoidf_(af[base + e0] + a_bias[d0]);
  float iclr1 = sigmoidf_(af[base + e1] + a_bias[d1]);
  float vv0 = sigmoidf_(bf[base + e0] + v_bias[d0]);
  float vv1 = sigmoidf_(bf[base + e1] + v_bias[d1]);
  float vr0 = vf[base + e0], vr1 = vf[base + e1];

  kf[base + e0] = kr0 * (1.0f + (iclr0 - 1.0f) * k_a[d0]);
  kf[base + e1] = kr1 * (1.0f + (iclr1 - 1.0f) * k_a[d1]);
  vf[base + e0] = vr0 + (v_first[base + e0] - vr0) * vv0;
  vf[base + e1] = vr1 + (v_first[base + e1] - vr1) * vv1;
  wf[base + e0] = __expf(-0.606531f * sigmoidf_(wf[base + e0] + w_bias[d0]));
  wf[base + e1] = __expf(-0.606531f * sigmoidf_(wf[base + e1] + w_bias[d1]));
  af[base + e0] = -kk0;
  af[base + e1] = -kk1;
  bf[base + e0] = kk0 * iclr0;
  bf[base + e1] = kk1 * iclr1;
}

// ---------------------------------------------------------------------------
// WKV7 scan. One block (256 threads) per (b,h). State rows are independent:
// 4 lanes share a row (16 f32 state columns each in registers); the row-dot
// reductions (sa . state, state . sr) use 2-step __shfl_xor inside the wave
// (lanes of a row differ only in bits [1:0]). Per-step vectors broadcast
// through LDS.
// ---------------------------------------------------------------------------
__global__ __launch_bounds__(256) void scan_kernel(
    const float* __restrict__ rf, const float* __restrict__ wf,
    const float* __restrict__ kf, const float* __restrict__ vf,
    const float* __restrict__ af, const float* __restrict__ bf,
    float* __restrict__ yf, int L, int H) {
  __shared__ float sr[64], sw[64], sk[64], sv[64], sa[64], sb[64];
  int b = blockIdx.x / H, h = blockIdx.x % H;
  int tid   = threadIdx.x;
  int row   = tid >> 2;        // 0..63
  int chunk = tid & 3;         // 0..3
  int c0    = chunk * 16;
  int e = tid & 63, g = tid >> 6;  // staging role

  float st[16];
#pragma unroll
  for (int j = 0; j < 16; ++j) st[j] = 0.0f;

  const size_t step = (size_t)H * 64;
  for (int t = 0; t < L; ++t) {
    size_t base = (((size_t)b * L + t) * H + h) * 64;
    if (g == 0)      { sr[e] = rf[base + e]; sw[e] = wf[base + e]; }
    else if (g == 1) { sk[e] = kf[base + e]; sv[e] = vf[base + e]; }
    else if (g == 2) { sa[e] = af[base + e];
                       __builtin_prefetch(af + base + step + e, 0, 1); }
    else             { sb[e] = bf[base + e];
                       __builtin_prefetch(rf + base + step + e, 0, 1); }
    __syncthreads();

    float ad = 0.0f;
#pragma unroll
    for (int j = 0; j < 16; ++j) ad += st[j] * sa[c0 + j];
    ad += __shfl_xor(ad, 1, 32);
    ad += __shfl_xor(ad, 2, 32);

    float vi = sv[row];
    float yv = 0.0f;
#pragma unroll
    for (int j = 0; j < 16; ++j) {
      st[j] = st[j] * sw[c0 + j] + vi * sk[c0 + j] + ad * sb[c0 + j];
      yv += st[j] * sr[c0 + j];
    }
    yv += __shfl_xor(yv, 1, 32);
    yv += __shfl_xor(yv, 2, 32);
    if (chunk == 0) yf[base + row] = yv;
    __syncthreads();
  }
}

// ---------------------------------------------------------------------------
// post: per-head groupnorm + (r.k*r_k)*v bonus + gate, emit bf16 operand for
// the final Wo GEMM. One wave per (b,l,h) vector.
// ---------------------------------------------------------------------------
__global__ __launch_bounds__(256) void post_kernel(
    const float* __restrict__ yf, const float* __restrict__ rf,
    const float* __restrict__ kf, const float* __restrict__ vf,
    const float* __restrict__ r_k, const float* __restrict__ gn_w,
    const float* __restrict__ gn_b, const float* __restrict__ gate,
    __bf16* __restrict__ z, int nvec, int H, int D) {
  int lane = threadIdx.x & 31;
  int vec  = blockIdx.x * 8 + (threadIdx.x >> 5);
  if (vec >= nvec) return;
  int h = vec & (H - 1);
  int token = vec / H;
  size_t base = (size_t)vec * 64;
  int e0 = lane, e1 = lane + 32;
  int d0 = h * 64 + e0, d1 = h * 64 + e1;

  float y0 = yf[base + e0], y1 = yf[base + e1];
  float mean = wave_sum(y0 + y1) * (1.0f / 64.0f);
  float var  = wave_sum(y0 * y0 + y1 * y1) * (1.0f / 64.0f) - mean * mean;
  float dot  = wave_sum(rf[base + e0] * kf[base + e0] * r_k[d0] +
                        rf[base + e1] * kf[base + e1] * r_k[d1]);
  float inv = rsqrtf(var + 0.00064f);
  float o0 = gn_w[d0] * (y0 - mean) * inv + gn_b[d0] + dot * vf[base + e0];
  float o1 = gn_w[d1] * (y1 - mean) * inv + gn_b[d1] + dot * vf[base + e1];
  size_t zi0 = (size_t)token * D + d0, zi1 = (size_t)token * D + d1;
  z[zi0] = f2bf(o0 * gate[zi0]);
  z[zi1] = f2bf(o1 * gate[zi1]);
}

// ---------------------------------------------------------------------------
// host-side orchestration
// ---------------------------------------------------------------------------
extern "C" void kernel_launch(void* const* d_in, const int* in_sizes, int n_in,
                              void* d_out, int out_size, void* d_ws, size_t ws_size,
                              hipStream_t stream) {
  (void)in_sizes; (void)n_in; (void)out_size; (void)ws_size;
  const int B = 4, L = 2048, D = 1024, H = 16;
  const int M = B * L;          // 8192 rows
  const int NVEC = B * L * H;   // 131072 head-vectors

  const float* x       = (const float*)d_in[0];
  const float* v_first = (const float*)d_in[1];
  const float* x_r = (const float*)d_in[2];
  const float* x_w = (const float*)d_in[3];
  const float* x_k = (const float*)d_in[4];
  const float* x_v = (const float*)d_in[5];
  const float* x_a = (const float*)d_in[6];
  const float* x_g = (const float*)d_in[7];
  const float* k_k = (const float*)d_in[8];
  const float* k_a = (const float*)d_in[9];
  const float* r_k = (const float*)d_in[10];
  const float* Wr  = (const float*)d_in[11];
  const float* Wk  = (const float*)d_in[12];
  const float* Wv  = (const float*)d_in[13];
  const float* Wo  = (const float*)d_in[14];
  const float* gn_w = (const float*)d_in[15];
  const float* gn_b = (const float*)d_in[16];
  const float* w_A = (const float*)d_in[17];
  const float* w_B = (const float*)d_in[18];
  const float* w_b = (const float*)d_in[19];
  const float* v_A = (const float*)d_in[20];
  const float* v_B = (const float*)d_in[21];
  const float* v_b = (const float*)d_in[22];
  const float* a_A = (const float*)d_in[23];
  const float* a_B = (const float*)d_in[24];
  const float* a_b = (const float*)d_in[25];
  const float* g_A = (const float*)d_in[26];
  const float* g_B = (const float*)d_in[27];

  char* ws = (char*)d_ws;
  size_t off = 0;
  auto alloc = [&](size_t bytes) -> void* {
    void* p = ws + off;
    off += (bytes + 255) & ~(size_t)255;
    return p;
  };

  const size_t MD = (size_t)M * D;
  __bf16* Wr_b = (__bf16*)alloc((size_t)D * D * 2);
  __bf16* Wk_b = (__bf16*)alloc((size_t)D * D * 2);
  __bf16* Wv_b = (__bf16*)alloc((size_t)D * D * 2);
  __bf16* Wo_b = (__bf16*)alloc((size_t)D * D * 2);
  __bf16* wA_b = (__bf16*)alloc((size_t)64 * D * 2);
  __bf16* wB_b = (__bf16*)alloc((size_t)D * 64 * 2);
  __bf16* vA_b = (__bf16*)alloc((size_t)32 * D * 2);
  __bf16* vB_b = (__bf16*)alloc((size_t)D * 32 * 2);
  __bf16* aA_b = (__bf16*)alloc((size_t)64 * D * 2);
  __bf16* aB_b = (__bf16*)alloc((size_t)D * 64 * 2);
  __bf16* gA_b = (__bf16*)alloc((size_t)128 * D * 2);
  __bf16* gB_b = (__bf16*)alloc((size_t)D * 128 * 2);

  __bf16* xr_b = (__bf16*)alloc(MD * 2);
  __bf16* xw_b = (__bf16*)alloc(MD * 2);
  __bf16* xk_b = (__bf16*)alloc(MD * 2);
  __bf16* xv_b = (__bf16*)alloc(MD * 2);
  __bf16* xa_b = (__bf16*)alloc(MD * 2);
  __bf16* xg_b = (__bf16*)alloc(MD * 2);

  float* r_f = (float*)alloc(MD * 4);   // r (stays)
  float* k_f = (float*)alloc(MD * 4);   // k raw -> k final (in place)
  float* v_f = (float*)alloc(MD * 4);   // v raw -> v final (in place)

  float* hw = (float*)alloc((size_t)M * 64 * 4);
  float* hv = (float*)alloc((size_t)M * 32 * 4);
  float* ha = (float*)alloc((size_t)M * 64 * 4);
  float* hg = (float*)alloc((size_t)M * 128 * 4);
  __bf16* hw_b = (__bf16*)alloc((size_t)M * 64 * 2);
  __bf16* hv_b = (__bf16*)alloc((size_t)M * 32 * 2);
  __bf16* ha_b = (__bf16*)alloc((size_t)M * 64 * 2);
  __bf16* hg_b = (__bf16*)alloc((size_t)M * 128 * 2);

  float* w_f    = (float*)alloc(MD * 4);  // w_pre -> w final (in place)
  float* a_f    = (float*)alloc(MD * 4);  // a_pre -> a = -kk (in place)
  float* bcoef  = (float*)alloc(MD * 4);  // v_mix -> b = kk*iclr (in place)
  float* gate_f = (float*)alloc(MD * 4);
  float* y_f    = (float*)alloc(MD * 4);
  __bf16* z_b   = (__bf16*)alloc(MD * 2);

  auto cvt = [&](const float* src, __bf16* dst, int n) {
    cvt_bf16_kernel<<<(n + 255) / 256, 256, 0, stream>>>(src, dst, n);
  };
  auto gemm = [&](const __bf16* A, const __bf16* Bm, float* C, int m, int n, int k) {
    if ((m % 256) == 0 && (n % 128) == 0) {
      dim3 grid(m / 256, n / 128);
      gemm_bf16_big_kernel<<<grid, 256, 0, stream>>>(A, Bm, C, m, n, k);
    } else {
      dim3 grid(m / 128, (n + 63) / 64);
      gemm_bf16_kernel<<<grid, 256, 0, stream>>>(A, Bm, C, m, n, k);
    }
  };

  // 1) weights -> bf16
  cvt(Wr, Wr_b, D * D);  cvt(Wk, Wk_b, D * D);
  cvt(Wv, Wv_b, D * D);  cvt(Wo, Wo_b, D * D);
  cvt(w_A, wA_b, 64 * D);  cvt(w_B, wB_b, D * 64);
  cvt(v_A, vA_b, 32 * D);  cvt(v_B, vB_b, D * 32);
  cvt(a_A, aA_b, 64 * D);  cvt(a_B, aB_b, D * 64);
  cvt(g_A, gA_b, 128 * D); cvt(g_B, gB_b, D * 128);

  // 2) token shift + mixes
  mix_kernel<<<(int)(MD / 256), 256, 0, stream>>>(
      x, x_r, x_w, x_k, x_v, x_a, x_g,
      xr_b, xw_b, xk_b, xv_b, xa_b, xg_b, (int)MD, D, L);

  // 3) main projections (big WMMA) + LoRA-down (small WMMA)
  gemm(xr_b, Wr_b, r_f, M, D, D);
  gemm(xk_b, Wk_b, k_f, M, D, D);
  gemm(xv_b, Wv_b, v_f, M, D, D);
  gemm(xw_b, wA_b, hw, M, 64, D);
  gemm(xv_b, vA_b, hv, M, 32, D);
  gemm(xa_b, aA_b, ha, M, 64, D);
  gemm(xg_b, gA_b, hg, M, 128, D);

  // 4) LoRA hidden activations -> bf16
  act_cvt_kernel<<<(M * 64 + 255) / 256, 256, 0, stream>>>(hw, hw_b, M * 64, 1);
  act_cvt_kernel<<<(M * 32 + 255) / 256, 256, 0, stream>>>(hv, hv_b, M * 32, 0);
  act_cvt_kernel<<<(M * 64 + 255) / 256, 256, 0, stream>>>(ha, ha_b, M * 64, 0);
  act_cvt_kernel<<<(M * 128 + 255) / 256, 256, 0, stream>>>(hg, hg_b, M * 128, 2);

  // 5) LoRA-up (big WMMA; N=1024, K = 64/32/64/128)
  gemm(hw_b, wB_b, w_f, M, D, 64);
  gemm(hv_b, vB_b, bcoef, M, D, 32);   // v_mix pre-sigmoid
  gemm(ha_b, aB_b, a_f, M, D, 64);     // a_pre
  gemm(hg_b, gB_b, gate_f, M, D, 128); // gate (sigmoid already inside)

  // 6) scan operand prep (in place)
  prep_kernel<<<NVEC / 8, 256, 0, stream>>>(
      k_f, v_f, w_f, a_f, bcoef, v_first, k_k, k_a, w_b, v_b, a_b, NVEC, H);

  // 7) sequential WKV7 scan (4 lanes per state row, 256 thr per (b,h))
  scan_kernel<<<B * H, 256, 0, stream>>>(r_f, w_f, k_f, v_f, a_f, bcoef,
                                         y_f, L, H);

  // 8) groupnorm + bonus + gate -> bf16
  post_kernel<<<NVEC / 8, 256, 0, stream>>>(
      y_f, r_f, k_f, v_f, r_k, gn_w, gn_b, gate_f, z_b, NVEC, H, D);

  // 9) output projection (big WMMA) -> d_out (f32)
  gemm(z_b, Wo_b, (float*)d_out, M, D, D);
}
